// PGCN_g_64845416235268
// MI455X (gfx1250) — matrix operand order repaired and checked by reading, not compiled, not run on hardware
//
#include <hip/hip_runtime.h>
#include <hip/hip_bf16.h>
#include <math.h>

#define N_NODES 50000
#define N_EDGES 800000
#define BGR     50
#define NPER    1000
#define KSEL    500

typedef __bf16 bf16_t;
typedef __attribute__((ext_vector_type(16))) __bf16 v16bf;
typedef __attribute__((ext_vector_type(8)))  float  v8f;

__device__ __forceinline__ void atomAddF(float* p, float v) {
    __hip_atomic_fetch_add(p, v, __ATOMIC_RELAXED, __HIP_MEMORY_SCOPE_AGENT);
}

// gfx1250 async VMEM->LDS copy (ASYNCcnt-tracked; no VGPR data round-trip)
__device__ __forceinline__ void async_ld_b128(unsigned lds_off, const void* gaddr) {
    asm volatile("global_load_async_to_lds_b128 %0, %1, off"
                 :: "v"(lds_off), "v"(gaddr) : "memory");
}
__device__ __forceinline__ void wait_async0() {
    asm volatile("s_wait_asynccnt 0" ::: "memory");
}

// ---------------------------------------------------------------- elementwise
__global__ void fill_f32(float* p, float v, int n) {
    int i = blockIdx.x * blockDim.x + threadIdx.x;
    if (i < n) p[i] = v;
}

__global__ void cast_f32_bf16(const float* __restrict__ in, bf16_t* __restrict__ out, int n) {
    int i = blockIdx.x * blockDim.x + threadIdx.x;
    if (i < n) out[i] = (bf16_t)in[i];
}

// pack W1A..D column-concat into [128, 512] bf16
__global__ void pack_w1(const float* __restrict__ A, const float* __restrict__ B,
                        const float* __restrict__ C, const float* __restrict__ D,
                        bf16_t* __restrict__ out) {
    int i = blockIdx.x * blockDim.x + threadIdx.x;  // 128*512
    if (i >= 128 * 512) return;
    int r = i >> 9, j = i & 511, q = j >> 7, c = j & 127;
    const float* W = (q == 0) ? A : (q == 1) ? B : (q == 2) ? C : D;
    out[i] = (bf16_t)W[r * 128 + c];
}

__global__ void pack_bias4(const float* a, const float* b, const float* c, const float* d,
                           float* out) {
    int i = blockIdx.x * blockDim.x + threadIdx.x;
    if (i >= 512) return;
    int q = i >> 7, j = i & 127;
    const float* B = (q == 0) ? a : (q == 1) ? b : (q == 2) ? c : d;
    out[i] = B[j];
}

// ---------------------------------------------------------------- degrees
__global__ void deg1_kernel(const int* __restrict__ ei, const float* __restrict__ ea,
                            float* __restrict__ deg4) {
    int e = blockIdx.x * blockDim.x + threadIdx.x;
    if (e >= N_EDGES) return;
    int dst = ei[N_EDGES + e];
    float4 w = ((const float4*)ea)[e];
    atomAddF(&deg4[0 * N_NODES + dst], w.x);
    atomAddF(&deg4[1 * N_NODES + dst], w.y);
    atomAddF(&deg4[2 * N_NODES + dst], w.z);
    atomAddF(&deg4[3 * N_NODES + dst], w.w);
}

__global__ void deg2_kernel(const int* __restrict__ ei, float* __restrict__ deg2) {
    int e = blockIdx.x * blockDim.x + threadIdx.x;
    if (e >= N_EDGES) return;
    atomAddF(&deg2[ei[N_EDGES + e]], 1.0f);
}

__global__ void rsqrt_kernel(float* p, int n) {
    int i = blockIdx.x * blockDim.x + threadIdx.x;
    if (i < n) { float d = p[i]; p[i] = (d > 0.0f) ? rsqrtf(d) : 0.0f; }
}

// ---------------------------------------------------------------- WMMA GEMM
// C[M,N](bf16) = A[M,K](bf16,row) @ B[K,N](bf16,row), f32 accumulate.
// 256 threads = 8 waves; BM=128, BN=64, BK=64; wave tile 32x32.
// Async VMEM->LDS double buffering: stage s+1 streams in while stage s computes.
#define SAS 72   // sA row stride in bf16 (64 + 8 pad)
#define SBS 72   // sB row stride in bf16 (64 + 8 pad)

__global__ void __launch_bounds__(256)
gemm_bf16_wmma(const bf16_t* __restrict__ A, const bf16_t* __restrict__ B,
               bf16_t* __restrict__ C, int M, int N, int K) {
    __shared__ __align__(16) bf16_t sA[2][128 * SAS];
    __shared__ __align__(16) bf16_t sB[2][64 * SBS];

    const int tid   = threadIdx.x;
    const int lane  = tid & 31;
    const int wave  = tid >> 5;
    const int waveM = wave & 3;   // 0..3 -> 32-row strips
    const int waveN = wave >> 2;  // 0..1 -> 32-col strips
    const int blockM = blockIdx.y * 128;
    const int blockN = blockIdx.x * 64;

    const int hf  = lane >> 4;    // half-wave id
    const int l16 = lane & 15;

    v8f acc[2][2] = {};

    union Frag { v16bf v; uint4 u[2]; };

    // per-thread staging geometry (constant across stages)
    const int ar = tid >> 1;                 // A row 0..127
    const int ac = (tid & 1) << 5;           // A col base 0 or 32
    int agr = blockM + ar;                   // clamp OOB rows: C rows are masked at store
    if (agr >= M) agr = M - 1;
    const int br = tid >> 2;                 // B row 0..63
    const int bc = (tid & 3) << 4;           // B col base 0,16,32,48

    auto stage = [&](int buf, int k0) {
        const bf16_t* ga = A + (size_t)agr * K + k0 + ac;
        unsigned la = (unsigned)(uintptr_t)(&sA[buf][ar * SAS + ac]);
        async_ld_b128(la,      ga);
        async_ld_b128(la + 16, ga + 8);
        async_ld_b128(la + 32, ga + 16);
        async_ld_b128(la + 48, ga + 24);
        const bf16_t* gb = B + (size_t)(k0 + br) * N + blockN + bc;
        unsigned lb = (unsigned)(uintptr_t)(&sB[buf][br * SBS + bc]);
        async_ld_b128(lb,      gb);
        async_ld_b128(lb + 16, gb + 8);
    };

    const int nStages = K >> 6;  // K multiple of 64 for all three GEMMs
    stage(0, 0);

    for (int s = 0; s < nStages; ++s) {
        const int buf = s & 1;
        wait_async0();       // this wave's tiles for stage s are in LDS
        __syncthreads();     // everyone's tiles are in LDS; prior reads of buf^1 done
        if (s + 1 < nStages) stage(buf ^ 1, (s + 1) << 6);

        #pragma unroll
        for (int kk = 0; kk < 64; kk += 32) {
            Frag af[2];
            #pragma unroll
            for (int sub = 0; sub < 2; ++sub) {
                const bf16_t* ap = &sA[buf][(waveM * 32 + sub * 16 + l16) * SAS + kk];
                af[sub].u[0] = *((const uint4*)(ap + 8 * hf));
                af[sub].u[1] = *((const uint4*)(ap + 16 + 8 * hf));
            }
            Frag bfg[2];
            #pragma unroll
            for (int sn = 0; sn < 2; ++sn) {
                const bf16_t* bp = &sB[buf][(kk + 8 * hf) * SBS + waveN * 32 + sn * 16 + l16];
                #pragma unroll
                for (int s = 0; s < 8; ++s) {
                    bfg[sn].v[s]     = bp[s * SBS];
                    bfg[sn].v[8 + s] = bp[(16 + s) * SBS];
                }
            }
            #pragma unroll
            for (int sub = 0; sub < 2; ++sub)
                #pragma unroll
                for (int sn = 0; sn < 2; ++sn)
                    acc[sub][sn] = __builtin_amdgcn_wmma_f32_16x16x32_bf16(
                        false, af[sub].v, false, bfg[sn].v,
                        (short)0, acc[sub][sn], false, false);
        }
    }

    // store: m = v + 8*hf, n = l16
    #pragma unroll
    for (int sub = 0; sub < 2; ++sub)
        #pragma unroll
        for (int sn = 0; sn < 2; ++sn) {
            int gm0 = blockM + waveM * 32 + sub * 16 + hf * 8;
            int gn  = blockN + waveN * 32 + sn * 16 + l16;
            #pragma unroll
            for (int v = 0; v < 8; ++v) {
                int gm = gm0 + v;
                if (gm < M) C[(size_t)gm * N + gn] = (bf16_t)acc[sub][sn][v];
            }
        }
}

// ---------------------------------------------------------------- scatter
__global__ void scatter_conv1(const int* __restrict__ ei, const float* __restrict__ ea,
                              const float* __restrict__ dinv4, const bf16_t* __restrict__ xw,
                              float* __restrict__ accv) {
    __shared__ float sc[4];
    __shared__ int s_src, s_dst;
    int e = blockIdx.x;
    if (threadIdx.x < 4) {
        int c = threadIdx.x;
        int src = ei[e], dst = ei[N_EDGES + e];
        if (c == 0) { s_src = src; s_dst = dst; }
        sc[c] = dinv4[c * N_NODES + src] * ea[(size_t)e * 4 + c] * dinv4[c * N_NODES + dst];
    }
    __syncthreads();
    int src = s_src, dst = s_dst, f = threadIdx.x;
    #pragma unroll
    for (int c = 0; c < 4; ++c) {
        float v = (float)xw[(size_t)src * 512 + c * 128 + f] * sc[c];
        atomAddF(&accv[(size_t)dst * 512 + c * 128 + f], v);
    }
}

__global__ void selfloop_conv1(const float* __restrict__ dinv4, const bf16_t* __restrict__ xw,
                               float* __restrict__ accv) {
    int i = blockIdx.x * blockDim.x + threadIdx.x;
    if (i >= N_NODES * 128) return;
    int n = i >> 7, f = i & 127;
    #pragma unroll
    for (int c = 0; c < 4; ++c) {
        float di = dinv4[c * N_NODES + n];
        size_t o = (size_t)n * 512 + c * 128 + f;
        accv[o] += (float)xw[o] * di * di;
    }
}

__global__ void scatter_ones(const int* __restrict__ ei, const float* __restrict__ dinv,
                             const bf16_t* __restrict__ xw, float* __restrict__ accv, int F2) {
    __shared__ float sc;
    __shared__ int s_src, s_dst;
    int e = blockIdx.x;
    if (threadIdx.x == 0) {
        int src = ei[e], dst = ei[N_EDGES + e];
        s_src = src; s_dst = dst;
        sc = dinv[src] * dinv[dst];
    }
    __syncthreads();
    int f = threadIdx.x;
    float v = (float)xw[(size_t)s_src * F2 + f] * sc;
    atomAddF(&accv[(size_t)s_dst * F2 + f], v);
}

__global__ void selfloop_ones(const float* __restrict__ dinv, const bf16_t* __restrict__ xw,
                              float* __restrict__ accv, int F2) {
    int i = blockIdx.x * blockDim.x + threadIdx.x;
    if (i >= N_NODES * F2) return;
    int n = i / F2;
    float di = dinv[n];
    accv[i] += (float)xw[i] * di * di;
}

__global__ void bias_relu_cast(float* __restrict__ accv, const float* __restrict__ bias,
                               bf16_t* __restrict__ obf, int F2, int do_relu, long total) {
    long i = (long)blockIdx.x * blockDim.x + threadIdx.x;
    if (i >= total) return;
    int j = (int)(i % F2);
    float v = accv[i] + bias[j];
    if (do_relu) v = fmaxf(v, 0.0f);
    accv[i] = v;
    if (obf) obf[i] = (bf16_t)v;
}

// ---------------------------------------------------------------- scoring
__global__ void score_node(const float* __restrict__ g1, const float* __restrict__ g2,
                           const float* __restrict__ g3, const float* __restrict__ w_rel,
                           const float* __restrict__ w_root,
                           float* __restrict__ s_rel, float* __restrict__ s_root) {
    int node = blockIdx.x * 8 + (threadIdx.x >> 5);
    if (node >= N_NODES) return;
    int lane = threadIdx.x & 31;
    float ar = 0.0f, ao = 0.0f;
    for (int j = lane; j < 896; j += 32) {
        float v;
        if (j < 512)      v = g1[(size_t)node * 512 + j];
        else if (j < 768) v = g2[(size_t)node * 256 + (j - 512)];
        else              v = g3[(size_t)node * 128 + (j - 768)];
        ar += v * w_rel[j];
        ao += v * w_root[j];
    }
    #pragma unroll
    for (int o = 16; o > 0; o >>= 1) {
        ar += __shfl_down(ar, o, 32);
        ao += __shfl_down(ao, o, 32);
    }
    if (lane == 0) { s_rel[node] = ar; s_root[node] = ao; }
}

__global__ void score_init(const float* __restrict__ s_root, const float* __restrict__ b_rel,
                           float* __restrict__ score) {
    int n = blockIdx.x * blockDim.x + threadIdx.x;
    if (n < N_NODES) score[n] = s_root[n] + b_rel[0];
}

__global__ void score_edges(const int* __restrict__ ei, const float* __restrict__ s_rel,
                            float* __restrict__ score) {
    int e = blockIdx.x * blockDim.x + threadIdx.x;
    if (e >= N_EDGES) return;
    atomAddF(&score[ei[N_EDGES + e]], s_rel[ei[e]]);
}

// ---------------------------------------------------------------- top-k + readout
__global__ void __launch_bounds__(256)
topk_readout(const float* __restrict__ score, const float* __restrict__ g1,
             const float* __restrict__ g2, const float* __restrict__ g3,
             float* __restrict__ readout) {
    __shared__ float ss[NPER];
    __shared__ float tv[NPER];
    __shared__ int   sf[NPER];
    int g = blockIdx.x;
    int tid = threadIdx.x;
    for (int i = tid; i < NPER; i += 256) ss[i] = score[g * NPER + i];
    __syncthreads();
    for (int i = tid; i < NPER; i += 256) {
        float si = ss[i];
        int rank = 0;
        for (int j = 0; j < NPER; ++j) {
            float sj = ss[j];
            rank += (sj > si) || (sj == si && j < i);
        }
        sf[i] = (rank < KSEL) ? 1 : 0;
        tv[i] = tanhf(si);
    }
    __syncthreads();
    for (int j = tid; j < 896; j += 256) {
        float sum = 0.0f, mx = -INFINITY;
        for (int i = 0; i < NPER; ++i) {
            if (!sf[i]) continue;
            size_t node = (size_t)g * NPER + i;
            float v;
            if (j < 512)      v = g1[node * 512 + j];
            else if (j < 768) v = g2[node * 256 + (j - 512)];
            else              v = g3[node * 128 + (j - 768)];
            v *= tv[i];
            sum += v;
            mx = fmaxf(mx, v);
        }
        readout[(size_t)g * 1792 + j]       = sum * (1.0f / KSEL);
        readout[(size_t)g * 1792 + 896 + j] = mx;
    }
}

// ---------------------------------------------------------------- MLP
__global__ void mlp1(const float* __restrict__ ro, const float* __restrict__ W,
                     const float* __restrict__ b, float* __restrict__ h1) {
    int pr = blockIdx.x * 8 + (threadIdx.x >> 5);   // (g,o) pair, 50*32=1600
    if (pr >= BGR * 32) return;
    int lane = threadIdx.x & 31;
    int g = pr >> 5, o = pr & 31;
    float a = 0.0f;
    for (int k = lane; k < 1792; k += 32)
        a += ro[(size_t)g * 1792 + k] * W[(size_t)k * 32 + o];
    #pragma unroll
    for (int s = 16; s > 0; s >>= 1) a += __shfl_down(a, s, 32);
    if (lane == 0) h1[g * 32 + o] = fmaxf(a + b[o], 0.0f);
}

__global__ void mlp2(const float* __restrict__ h1, const float* __restrict__ W,
                     const float* __restrict__ b, float* __restrict__ h2) {
    int i = blockIdx.x * blockDim.x + threadIdx.x;  // 50*8
    if (i >= BGR * 8) return;
    int g = i >> 3, o = i & 7;
    float a = 0.0f;
    for (int k = 0; k < 32; ++k) a += h1[g * 32 + k] * W[k * 8 + o];
    h2[i] = fmaxf(a + b[o], 0.0f);
}

__global__ void mlp3(const float* __restrict__ h2, const float* __restrict__ W,
                     const float* __restrict__ b, float* __restrict__ out) {
    int i = blockIdx.x * blockDim.x + threadIdx.x;  // 50*2
    if (i >= BGR * 2) return;
    int g = i >> 1, o = i & 1;
    float a = 0.0f;
    for (int k = 0; k < 8; ++k) a += h2[g * 8 + k] * W[k * 2 + o];
    out[i] = a + b[o];
}

// ---------------------------------------------------------------- launch
extern "C" void kernel_launch(void* const* d_in, const int* in_sizes, int n_in,
                              void* d_out, int out_size, void* d_ws, size_t ws_size,
                              hipStream_t stream) {
    (void)in_sizes; (void)n_in; (void)out_size; (void)ws_size;
    const float* x     = (const float*)d_in[0];
    const int*   ei    = (const int*)  d_in[1];
    const float* ea    = (const float*)d_in[2];
    const float* W1A = (const float*)d_in[4];  const float* b1A = (const float*)d_in[5];
    const float* W1B = (const float*)d_in[6];  const float* b1B = (const float*)d_in[7];
    const float* W1C = (const float*)d_in[8];  const float* b1C = (const float*)d_in[9];
    const float* W1D = (const float*)d_in[10]; const float* b1D = (const float*)d_in[11];
    const float* W2  = (const float*)d_in[12]; const float* b2  = (const float*)d_in[13];
    const float* W3  = (const float*)d_in[14]; const float* b3  = (const float*)d_in[15];
    const float* w_rel = (const float*)d_in[16]; const float* b_rel = (const float*)d_in[17];
    const float* w_root= (const float*)d_in[18];
    const float* Wm1 = (const float*)d_in[19]; const float* bm1 = (const float*)d_in[20];
    const float* Wm2 = (const float*)d_in[21]; const float* bm2 = (const float*)d_in[22];
    const float* Wm3 = (const float*)d_in[23]; const float* bm3 = (const float*)d_in[24];
    float* out = (float*)d_out;

    char* ws = (char*)d_ws;
    size_t off = 0;
    auto alloc = [&](size_t bytes) -> char* {
        char* p = ws + off;
        off += (bytes + 255) & ~(size_t)255;
        return p;
    };

    const size_t N = N_NODES;
    float*  gcn1   = (float*) alloc(N * 512 * 4);
    float*  gcn2   = (float*) alloc(N * 256 * 4);
    float*  gcn3   = (float*) alloc(N * 128 * 4);
    bf16_t* xwbf   = (bf16_t*)alloc(N * 512 * 2);   // reused: xw1/xw2/xw3
    bf16_t* abf    = (bf16_t*)alloc(N * 512 * 2);   // reused: x_bf / gcn1_bf / gcn2_bf
    bf16_t* wpack  = (bf16_t*)alloc(512 * 512 * 2);
    float*  bpack  = (float*) alloc(512 * 4);
    float*  deg4   = (float*) alloc(4 * N * 4);
    float*  deg2   = (float*) alloc(N * 4);
    float*  s_rel  = (float*) alloc(N * 4);
    float*  s_root = (float*) alloc(N * 4);
    float*  score  = (float*) alloc(N * 4);
    float*  rdout  = (float*) alloc((size_t)BGR * 1792 * 4);
    float*  h1     = (float*) alloc(BGR * 32 * 4);
    float*  h2     = (float*) alloc(BGR * 8 * 4);

    auto blk = [](long n) { return dim3((unsigned)((n + 255) / 256)); };

    // ---- degrees (self-loop weight 1 folded into init)
    fill_f32<<<blk(4 * N), 256, 0, stream>>>(deg4, 1.0f, 4 * N);
    fill_f32<<<blk(N), 256, 0, stream>>>(deg2, 1.0f, N);
    deg1_kernel<<<blk(N_EDGES), 256, 0, stream>>>(ei, ea, deg4);
    deg2_kernel<<<blk(N_EDGES), 256, 0, stream>>>(ei, deg2);
    rsqrt_kernel<<<blk(4 * N), 256, 0, stream>>>(deg4, 4 * N);
    rsqrt_kernel<<<blk(N), 256, 0, stream>>>(deg2, N);

    // ---- conv1: X(bf16) @ [W1A|W1B|W1C|W1D]
    cast_f32_bf16<<<blk(N * 128), 256, 0, stream>>>(x, abf, N * 128);
    pack_w1<<<blk(128 * 512), 256, 0, stream>>>(W1A, W1B, W1C, W1D, wpack);
    pack_bias4<<<2, 256, 0, stream>>>(b1A, b1B, b1C, b1D, bpack);
    {
        dim3 grid(512 / 64, (N_NODES + 127) / 128);
        gemm_bf16_wmma<<<grid, 256, 0, stream>>>(abf, wpack, xwbf, N_NODES, 512, 128);
    }
    hipMemsetAsync(gcn1, 0, N * 512 * 4, stream);
    scatter_conv1<<<N_EDGES, 128, 0, stream>>>(ei, ea, deg4, xwbf, gcn1);
    selfloop_conv1<<<blk(N * 128), 256, 0, stream>>>(deg4, xwbf, gcn1);
    bias_relu_cast<<<blk(N * 512), 256, 0, stream>>>(gcn1, bpack, abf, 512, 1, (long)N * 512);

    // ---- conv2: gcn1(bf16) @ W2
    cast_f32_bf16<<<blk(512 * 256), 256, 0, stream>>>(W2, wpack, 512 * 256);
    {
        dim3 grid(256 / 64, (N_NODES + 127) / 128);
        gemm_bf16_wmma<<<grid, 256, 0, stream>>>(abf, wpack, xwbf, N_NODES, 256, 512);
    }
    hipMemsetAsync(gcn2, 0, N * 256 * 4, stream);
    scatter_ones<<<N_EDGES, 256, 0, stream>>>(ei, deg2, xwbf, gcn2, 256);
    selfloop_ones<<<blk(N * 256), 256, 0, stream>>>(deg2, xwbf, gcn2, 256);
    bias_relu_cast<<<blk(N * 256), 256, 0, stream>>>(gcn2, b2, abf, 256, 1, (long)N * 256);

    // ---- conv3: gcn2(bf16) @ W3 (no relu)
    cast_f32_bf16<<<blk(256 * 128), 256, 0, stream>>>(W3, wpack, 256 * 128);
    {
        dim3 grid(128 / 64, (N_NODES + 127) / 128);
        gemm_bf16_wmma<<<grid, 256, 0, stream>>>(abf, wpack, xwbf, N_NODES, 128, 256);
    }
    hipMemsetAsync(gcn3, 0, N * 128 * 4, stream);
    scatter_ones<<<N_EDGES, 128, 0, stream>>>(ei, deg2, xwbf, gcn3, 128);
    selfloop_ones<<<blk(N * 128), 256, 0, stream>>>(deg2, xwbf, gcn3, 128);
    bias_relu_cast<<<blk(N * 128), 256, 0, stream>>>(gcn3, b3, (bf16_t*)nullptr, 128, 0, (long)N * 128);

    // ---- SAGPool score: per-node scalars first, then scalar edge scatter
    score_node<<<(N_NODES + 7) / 8, 256, 0, stream>>>(gcn1, gcn2, gcn3, w_rel, w_root, s_rel, s_root);
    score_init<<<blk(N), 256, 0, stream>>>(s_root, b_rel, score);
    score_edges<<<blk(N_EDGES), 256, 0, stream>>>(ei, s_rel, score);

    // ---- top-k selection + mean/max readout (order-invariant)
    topk_readout<<<BGR, 256, 0, stream>>>(score, gcn1, gcn2, gcn3, rdout);

    // ---- MLP head
    mlp1<<<(BGR * 32) / 8, 256, 0, stream>>>(rdout, Wm1, bm1, h1);
    mlp2<<<blk(BGR * 8), 256, 0, stream>>>(h1, Wm2, bm2, h2);
    mlp3<<<blk(BGR * 2), 256, 0, stream>>>(h2, Wm3, bm3, out);
}
